// BidirectionalCrossAttention_26706106646680
// MI455X (gfx1250) — compile-verified, matrix operand order checked by reading, not run on hardware
//
#include <hip/hip_runtime.h>
#include <math.h>

typedef __bf16 bf16_t;
typedef __attribute__((ext_vector_type(16))) __bf16 v16bf;
typedef __attribute__((ext_vector_type(8)))  __bf16 v8bf;
typedef __attribute__((ext_vector_type(8)))  float  v8f;

// ---------- helpers ----------
__device__ inline float bf2f(bf16_t x) {
    unsigned short u = __builtin_bit_cast(unsigned short, x);
    unsigned int w = ((unsigned int)u) << 16;
    return __builtin_bit_cast(float, w);
}
__device__ inline bf16_t f2bf(float x) {
    unsigned int w = __builtin_bit_cast(unsigned int, x);
    unsigned int r = (w + 0x7FFFu + ((w >> 16) & 1u)) >> 16;
    return __builtin_bit_cast(bf16_t, (unsigned short)r);
}
__device__ inline float wave_sum(float v) {
    #pragma unroll
    for (int off = 16; off > 0; off >>= 1) v += __shfl_xor(v, off, 32);
    return v;
}
__device__ inline float sigmoidf_(float x) { return 1.f / (1.f + expf(-x)); }

// Load one 16x32 bf16 fragment (A-layout per CDNA5 ISA 7.12.2) from row-major src.
// lane<16: M=lane, K={koff0..+7, koff0+16..+23} with koff0=0 ; lane>=16: koff0=8.
__device__ inline v16bf load_frag16x32(const bf16_t* src, int ld, int lane) {
    int m = lane & 15;
    int koff = (lane & 16) ? 8 : 0;
    const bf16_t* p = src + (size_t)m * ld + koff;
    union { v16bf v; v8bf h[2]; } f;
    f.h[0] = *(const v8bf*)(p);
    f.h[1] = *(const v8bf*)(p + 16);
    return f.v;
}
__device__ inline v8f wmma_bf16(v16bf a, v16bf b, v8f c) {
    return __builtin_amdgcn_wmma_f32_16x16x32_bf16(false, a, false, b, (short)0, c, false, false);
}
__device__ inline void load16bf(const bf16_t* p, float* dst) {
    v8bf a = *(const v8bf*)p, b = *(const v8bf*)(p + 8);
    #pragma unroll
    for (int i = 0; i < 8; ++i) { dst[i] = bf2f(a[i]); dst[8 + i] = bf2f(b[i]); }
}
__device__ inline void store16bf(bf16_t* p, const float* src) {
    v8bf a, b;
    #pragma unroll
    for (int i = 0; i < 8; ++i) { a[i] = f2bf(src[i]); b[i] = f2bf(src[8 + i]); }
    *(v8bf*)p = a; *(v8bf*)(p + 8) = b;
}

// ---------- weight transpose/convert: Wt[n*K+k] = bf16(W[k*N+n]) ----------
__global__ void k_transpose_bf16(const float* __restrict__ W, bf16_t* __restrict__ Wt, int K, int N) {
    int idx = blockIdx.x * 256 + threadIdx.x;
    if (idx >= K * N) return;
    int n = idx / K, k = idx - n * K;
    Wt[idx] = f2bf(W[(size_t)k * N + n]);
}

// ---------- LN over rgb, transpose to grouped (pixel-unshuffled) token order ----------
__global__ __launch_bounds__(256) void k_ln_rgb1(const float* __restrict__ rgb,
                                                 const float* __restrict__ g, const float* __restrict__ b_,
                                                 bf16_t* __restrict__ out) {
    __shared__ float sm[128][33];
    int bi = blockIdx.x; int wch = bi & 7; int h = (bi >> 3) & 255; int b = bi >> 11;
    int tid = threadIdx.x, lane = tid & 31, wv = tid >> 5;
    int w0 = wch * 32;
    #pragma unroll
    for (int it = 0; it < 16; ++it) {
        int c = it * 8 + wv;
        sm[c][lane] = rgb[(((size_t)b * 128 + c) * 256 + h) * 256 + w0 + lane];
    }
    __syncthreads();
    int hc = h >> 2, dh = h & 3;
    for (int ti = 0; ti < 4; ++ti) {
        int wl = wv * 4 + ti;
        float x[4], s = 0.f;
        #pragma unroll
        for (int j = 0; j < 4; ++j) { x[j] = sm[lane + 32 * j][wl]; s += x[j]; }
        s = wave_sum(s); float mu = s * (1.f / 128.f);
        float q = 0.f;
        #pragma unroll
        for (int j = 0; j < 4; ++j) { float d = x[j] - mu; q += d * d; }
        q = wave_sum(q); float rs = rsqrtf(q * (1.f / 128.f) + 1e-5f);
        int w = w0 + wl, wc = w >> 2, dw = w & 3;
        size_t gidx = (((size_t)((b * 64 + hc) * 64 + wc)) * 16 + dh * 4 + dw) * 128;
        #pragma unroll
        for (int j = 0; j < 4; ++j) { int c = lane + 32 * j; out[gidx + c] = f2bf((x[j] - mu) * rs * g[c] + b_[c]); }
    }
}

// ---------- LN over hsi, transpose to token order (loc*31+s) ----------
__global__ __launch_bounds__(256) void k_ln_hsi1(const float* __restrict__ hsi,
                                                 const float* __restrict__ g, const float* __restrict__ b_,
                                                 bf16_t* __restrict__ out) {
    __shared__ float sm[128][33];
    int bi = blockIdx.x;
    int wch = bi & 1; int t = bi >> 1; int hc = t & 63; t >>= 6; int s = t % 31; int b = t / 31;
    int tid = threadIdx.x, lane = tid & 31, wv = tid >> 5;
    #pragma unroll
    for (int it = 0; it < 16; ++it) {
        int c = it * 8 + wv;
        sm[c][lane] = hsi[(((size_t)b * 128 + c) * 31 + s) * 4096 + hc * 64 + wch * 32 + lane];
    }
    __syncthreads();
    for (int ti = 0; ti < 4; ++ti) {
        int wl = wv * 4 + ti;
        float x[4], su = 0.f;
        #pragma unroll
        for (int j = 0; j < 4; ++j) { x[j] = sm[lane + 32 * j][wl]; su += x[j]; }
        su = wave_sum(su); float mu = su * (1.f / 128.f);
        float q = 0.f;
        #pragma unroll
        for (int j = 0; j < 4; ++j) { float d = x[j] - mu; q += d * d; }
        q = wave_sum(q); float rs = rsqrtf(q * (1.f / 128.f) + 1e-5f);
        int wc = wch * 32 + wl;
        size_t tok = ((size_t)((b * 64 + hc) * 64 + wc)) * 31 + s;
        #pragma unroll
        for (int j = 0; j < 4; ++j) { int c = lane + 32 * j; out[tok * 128 + c] = f2bf((x[j] - mu) * rs * g[c] + b_[c]); }
    }
}

// ---------- GEMM: Y[N,128] = X[N,128] @ Wt^T + bias  (bf16 in, bf16 out) ----------
__global__ __launch_bounds__(256) void k_gemm_bias(const bf16_t* __restrict__ X, const bf16_t* __restrict__ Wt,
                                                   const float* __restrict__ bias, bf16_t* __restrict__ Y) {
    int lane = threadIdx.x & 31, wv = threadIdx.x >> 5;
    const bf16_t* Xb = X + (size_t)blockIdx.x * 16 * 128;
    const bf16_t* Wb = Wt + (size_t)wv * 16 * 128;
    v8f acc = {};
    #pragma unroll
    for (int kb = 0; kb < 4; ++kb) {
        v16bf a = load_frag16x32(Xb + kb * 32, 128, lane);
        v16bf bb = load_frag16x32(Wb + kb * 32, 128, lane);
        acc = wmma_bf16(a, bb, acc);
    }
    int n = wv * 16 + (lane & 15);
    int mb = (lane & 16) ? 8 : 0;
    float bn = bias[n];
    #pragma unroll
    for (int v = 0; v < 8; ++v) {
        int m = v + mb;
        Y[((size_t)blockIdx.x * 16 + m) * 128 + n] = f2bf(acc[v] + bn);
    }
}

// ---------- proj + residual (rgb side, grouped tokens), f32 out ----------
__global__ __launch_bounds__(256) void k_proj_res_rgb(const bf16_t* __restrict__ X, const bf16_t* __restrict__ Wt,
                                                      const float* __restrict__ bias, const float* __restrict__ rgb,
                                                      float* __restrict__ Z) {
    int lane = threadIdx.x & 31, wv = threadIdx.x >> 5;
    const bf16_t* Xb = X + (size_t)blockIdx.x * 16 * 128;
    const bf16_t* Wb = Wt + (size_t)wv * 16 * 128;
    v8f acc = {};
    #pragma unroll
    for (int kb = 0; kb < 4; ++kb)
        acc = wmma_bf16(load_frag16x32(Xb + kb * 32, 128, lane), load_frag16x32(Wb + kb * 32, 128, lane), acc);
    int n = wv * 16 + (lane & 15);
    int mb = (lane & 16) ? 8 : 0;
    float bn = bias[n];
    #pragma unroll
    for (int v = 0; v < 8; ++v) {
        size_t tok = (size_t)blockIdx.x * 16 + v + mb;
        int child = (int)(tok & 15); size_t loc = tok >> 4;
        int b = (int)(loc >> 12), hc = (int)((loc >> 6) & 63), wc = (int)(loc & 63);
        int h = hc * 4 + (child >> 2), w = wc * 4 + (child & 3);
        float resid = rgb[(((size_t)b * 128 + n) * 256 + h) * 256 + w];
        Z[tok * 128 + n] = acc[v] + bn + resid;
    }
}

// ---------- proj + residual (hsi side), f32 out ----------
__global__ __launch_bounds__(256) void k_proj_res_hsi(const bf16_t* __restrict__ X, const bf16_t* __restrict__ Wt,
                                                      const float* __restrict__ bias, const float* __restrict__ hsi,
                                                      float* __restrict__ Z) {
    int lane = threadIdx.x & 31, wv = threadIdx.x >> 5;
    const bf16_t* Xb = X + (size_t)blockIdx.x * 16 * 128;
    const bf16_t* Wb = Wt + (size_t)wv * 16 * 128;
    v8f acc = {};
    #pragma unroll
    for (int kb = 0; kb < 4; ++kb)
        acc = wmma_bf16(load_frag16x32(Xb + kb * 32, 128, lane), load_frag16x32(Wb + kb * 32, 128, lane), acc);
    int n = wv * 16 + (lane & 15);
    int mb = (lane & 16) ? 8 : 0;
    float bn = bias[n];
    #pragma unroll
    for (int v = 0; v < 8; ++v) {
        size_t tok = (size_t)blockIdx.x * 16 + v + mb;
        size_t loc = tok / 31; int s = (int)(tok - loc * 31);
        int b = (int)(loc >> 12), hc = (int)((loc >> 6) & 63), wc = (int)(loc & 63);
        float resid = hsi[(((size_t)b * 128 + n) * 31 + s) * 4096 + hc * 64 + wc];
        Z[tok * 128 + n] = acc[v] + bn + resid;
    }
}

// ---------- attention RGB->HSI: 16 child queries x 31 keys per location, wave = head ----------
__global__ __launch_bounds__(256) void k_attn_r2h(const bf16_t* __restrict__ Q, const bf16_t* __restrict__ K,
                                                  const bf16_t* __restrict__ V, bf16_t* __restrict__ O) {
    size_t loc = blockIdx.x;
    int hd = threadIdx.x >> 5, lane = threadIdx.x & 31;
    int qi = lane & 15; bool hi = lane >= 16;
    float qv[16];
    load16bf(Q + ((loc * 16 + qi) * 128 + hd * 16), qv);
    int k0 = hi ? 16 : 0, kn = hi ? 15 : 16;
    float sc[16]; float mx = -3.0e38f;
    for (int i = 0; i < kn; ++i) {
        float kvv[16];
        load16bf(K + ((loc * 31 + k0 + i) * 128 + hd * 16), kvv);
        float s = 0.f;
        #pragma unroll
        for (int d = 0; d < 16; ++d) s += qv[d] * kvv[d];
        s *= 0.25f; sc[i] = s; mx = fmaxf(mx, s);
    }
    float gm = fmaxf(mx, __shfl_xor(mx, 16, 32));
    float sm = 0.f;
    for (int i = 0; i < kn; ++i) { sc[i] = expf(sc[i] - gm); sm += sc[i]; }
    float gsum = sm + __shfl_xor(sm, 16, 32);
    float inv = 1.f / gsum;
    float acc[16];
    #pragma unroll
    for (int d = 0; d < 16; ++d) acc[d] = 0.f;
    for (int i = 0; i < kn; ++i) {
        float vv[16];
        load16bf(V + ((loc * 31 + k0 + i) * 128 + hd * 16), vv);
        #pragma unroll
        for (int d = 0; d < 16; ++d) acc[d] += sc[i] * vv[d];
    }
    #pragma unroll
    for (int d = 0; d < 16; ++d) acc[d] = (acc[d] + __shfl_xor(acc[d], 16, 32)) * inv;
    if (!hi) store16bf(O + ((loc * 16 + qi) * 128 + hd * 16), acc);
}

// ---------- attention HSI->RGB: 31 spectral queries x 16 child keys, wave = head ----------
__global__ __launch_bounds__(256) void k_attn_h2r(const bf16_t* __restrict__ Q, const bf16_t* __restrict__ K,
                                                  const bf16_t* __restrict__ V, bf16_t* __restrict__ O) {
    size_t loc = blockIdx.x;
    int hd = threadIdx.x >> 5, lane = threadIdx.x & 31;
    bool valid = lane < 31;
    int s = valid ? lane : 30;
    float qv[16];
    load16bf(Q + ((loc * 31 + s) * 128 + hd * 16), qv);
    float sc[16]; float mx = -3.0e38f;
    #pragma unroll
    for (int k = 0; k < 16; ++k) {
        float kvv[16];
        load16bf(K + ((loc * 16 + k) * 128 + hd * 16), kvv);
        float t = 0.f;
        #pragma unroll
        for (int d = 0; d < 16; ++d) t += qv[d] * kvv[d];
        t *= 0.25f; sc[k] = t; mx = fmaxf(mx, t);
    }
    float sm = 0.f;
    #pragma unroll
    for (int k = 0; k < 16; ++k) { sc[k] = expf(sc[k] - mx); sm += sc[k]; }
    float inv = 1.f / sm;
    float acc[16];
    #pragma unroll
    for (int d = 0; d < 16; ++d) acc[d] = 0.f;
    #pragma unroll
    for (int k = 0; k < 16; ++k) {
        float vv[16];
        load16bf(V + ((loc * 16 + k) * 128 + hd * 16), vv);
        #pragma unroll
        for (int d = 0; d < 16; ++d) acc[d] += sc[k] * vv[d];
    }
    #pragma unroll
    for (int d = 0; d < 16; ++d) acc[d] *= inv;
    if (valid) store16bf(O + ((loc * 31 + s) * 128 + hd * 16), acc);
}

// ---------- fused LN + FFN (128->512 gelu -> 128) + residual, in place on f32 tokens ----------
__global__ __launch_bounds__(128) void k_ffn_fused(float* __restrict__ Z, const float* __restrict__ g,
                                                   const float* __restrict__ b_, const bf16_t* __restrict__ W1t,
                                                   const float* __restrict__ b1, const bf16_t* __restrict__ W2t,
                                                   const float* __restrict__ b2) {
    __shared__ bf16_t Xs[16 * 128];
    __shared__ bf16_t Hs[16 * 512];
    __shared__ float red[16][8][2];
    __shared__ float stat[16][2];
    int tid = threadIdx.x, lane = tid & 31, wv = tid >> 5;
    size_t t0 = (size_t)blockIdx.x * 16;
    // phase 0: LayerNorm -> Xs (bf16)
    int token = tid >> 3, part = tid & 7;
    float xv[16]; float s = 0.f, q = 0.f;
    const float* xp = Z + (t0 + token) * 128 + part * 16;
    #pragma unroll
    for (int i = 0; i < 16; ++i) { xv[i] = xp[i]; s += xv[i]; q += xv[i] * xv[i]; }
    red[token][part][0] = s; red[token][part][1] = q;
    __syncthreads();
    if (part == 0) {
        float ss = 0.f, qq = 0.f;
        #pragma unroll
        for (int p = 0; p < 8; ++p) { ss += red[token][p][0]; qq += red[token][p][1]; }
        float mu = ss * (1.f / 128.f);
        stat[token][0] = mu;
        stat[token][1] = rsqrtf(qq * (1.f / 128.f) - mu * mu + 1e-5f);
    }
    __syncthreads();
    {
        float mu = stat[token][0], rs = stat[token][1];
        #pragma unroll
        for (int i = 0; i < 16; ++i) { int c = part * 16 + i; Xs[token * 128 + c] = f2bf((xv[i] - mu) * rs * g[c] + b_[c]); }
    }
    __syncthreads();
    // phase 1: hidden = gelu(X @ W1 + b1)  -> Hs
    for (int ct = 0; ct < 8; ++ct) {
        int cb = wv * 128 + ct * 16;
        v8f acc = {};
        #pragma unroll
        for (int kb = 0; kb < 4; ++kb)
            acc = wmma_bf16(load_frag16x32(Xs + kb * 32, 128, lane),
                            load_frag16x32(W1t + (size_t)cb * 128 + kb * 32, 128, lane), acc);
        int n = cb + (lane & 15), mb = (lane & 16) ? 8 : 0;
        float bn = b1[n];
        #pragma unroll
        for (int v = 0; v < 8; ++v) {
            int m = v + mb;
            float x = acc[v] + bn;
            float gx = 0.5f * x * (1.f + erff(x * 0.70710678118f));
            Hs[m * 512 + n] = f2bf(gx);
        }
    }
    __syncthreads();
    // phase 2: out = H @ W2 + b2 + residual, in place
    for (int ct = 0; ct < 2; ++ct) {
        int cb = wv * 32 + ct * 16;
        v8f acc = {};
        #pragma unroll
        for (int kb = 0; kb < 16; ++kb)
            acc = wmma_bf16(load_frag16x32(Hs + kb * 32, 512, lane),
                            load_frag16x32(W2t + (size_t)cb * 512 + kb * 32, 512, lane), acc);
        int n = cb + (lane & 15), mb = (lane & 16) ? 8 : 0;
        float bn = b2[n];
        #pragma unroll
        for (int v = 0; v < 8; ++v) {
            int m = v + mb;
            size_t idx = (t0 + m) * 128 + n;
            Z[idx] = acc[v] + bn + Z[idx];
        }
    }
}

// ---------- SE pooling: per 512-token chunk per-channel sums (deterministic two-pass) ----------
__global__ __launch_bounds__(128) void k_se_pool(const float* __restrict__ Z, float* __restrict__ partials) {
    int c = threadIdx.x;
    size_t ch = blockIdx.x;
    const float* p = Z + ch * 512 * 128 + c;
    float s = 0.f;
    for (int i = 0; i < 512; ++i) s += p[(size_t)i * 128];
    partials[ch * 128 + c] = s;
}

__global__ __launch_bounds__(128) void k_se_mlp(const float* __restrict__ partials, const float* __restrict__ W1,
                                                const float* __restrict__ b1, const float* __restrict__ W2,
                                                const float* __restrict__ b2, float* __restrict__ wbuf) {
    __shared__ float pooled[2][128];
    __shared__ float hid[2][16];
    int c = threadIdx.x;
    for (int b = 0; b < 2; ++b) {
        float s = 0.f;
        for (int ch = 0; ch < 248; ++ch) s += partials[((size_t)b * 248 + ch) * 128 + c];
        pooled[b][c] = s * (1.f / 126976.f);
    }
    __syncthreads();
    if (c < 32) {
        int j = c & 15, b = c >> 4;
        float s = b1[j];
        for (int cc = 0; cc < 128; ++cc) s += pooled[b][cc] * W1[cc * 16 + j];
        hid[b][j] = fmaxf(s, 0.f);
    }
    __syncthreads();
    for (int b = 0; b < 2; ++b) {
        float s = b2[c];
        #pragma unroll
        for (int j = 0; j < 16; ++j) s += hid[b][j] * W2[j * 128 + c];
        wbuf[b * 128 + c] = sigmoidf_(s);
    }
}

// ---------- spectral collapse + LN(ln_hsi_fuse) -> FUS[loc][c] ----------
__global__ __launch_bounds__(256) void k_collapse(const float* __restrict__ Zh, const float* __restrict__ wbuf,
                                                  const float* __restrict__ g, const float* __restrict__ b_,
                                                  float* __restrict__ FUS) {
    int lane = threadIdx.x & 31, wv = threadIdx.x >> 5;
    size_t loc = (size_t)blockIdx.x * 8 + wv;
    int b = (int)(loc >> 12);
    float x[4]; float s = 0.f;
    #pragma unroll
    for (int j = 0; j < 4; ++j) {
        int c = lane + 32 * j;
        float a = 0.f;
        const float* p = Zh + (loc * 31) * 128 + c;
        for (int ss = 0; ss < 31; ++ss) a += p[(size_t)ss * 128];
        x[j] = wbuf[b * 128 + c] * a * (1.f / 31.f);
        s += x[j];
    }
    s = wave_sum(s); float mu = s * (1.f / 128.f);
    float q = 0.f;
    #pragma unroll
    for (int j = 0; j < 4; ++j) { float d = x[j] - mu; q += d * d; }
    q = wave_sum(q); float rs = rsqrtf(q * (1.f / 128.f) + 1e-5f);
    #pragma unroll
    for (int j = 0; j < 4; ++j) { int c = lane + 32 * j; FUS[loc * 128 + c] = (x[j] - mu) * rs * g[c] + b_[c]; }
}

// ---------- final: LN(rgb_fuse) + gate*up, LN(fuse), transpose to (B,C,H,W) ----------
__global__ __launch_bounds__(256) void k_final(const float* __restrict__ Zr, const float* __restrict__ FUS,
                                               const float* __restrict__ alpha, const float* __restrict__ gR,
                                               const float* __restrict__ bR, const float* __restrict__ gF,
                                               const float* __restrict__ bF, float* __restrict__ out) {
    __shared__ float tile[128][33];
    int tid = threadIdx.x, lane = tid & 31, wv = tid >> 5;
    int bi = blockIdx.x; int wch = bi & 7; int h = (bi >> 3) & 255; int b = bi >> 11;
    int hc = h >> 2, dh = h & 3;
    for (int ti = 0; ti < 4; ++ti) {
        int wl = wv * 4 + ti;
        int w = wch * 32 + wl, wc = w >> 2, dw = w & 3;
        size_t loc = ((size_t)(b * 64 + hc)) * 64 + wc;
        size_t gidx = (loc * 16 + dh * 4 + dw) * 128;
        float x[4]; float s = 0.f;
        #pragma unroll
        for (int j = 0; j < 4; ++j) { x[j] = Zr[gidx + lane + 32 * j]; s += x[j]; }
        s = wave_sum(s); float mu = s * (1.f / 128.f);
        float q = 0.f;
        #pragma unroll
        for (int j = 0; j < 4; ++j) { float d = x[j] - mu; q += d * d; }
        q = wave_sum(q); float rs = rsqrtf(q * (1.f / 128.f) + 1e-5f);
        float y[4]; float s2 = 0.f;
        #pragma unroll
        for (int j = 0; j < 4; ++j) {
            int c = lane + 32 * j;
            float rf = (x[j] - mu) * rs * gR[c] + bR[c];
            float gate = sigmoidf_(alpha[c]);
            y[j] = rf + gate * FUS[loc * 128 + c];
            s2 += y[j];
        }
        s2 = wave_sum(s2); float mu2 = s2 * (1.f / 128.f);
        float q2 = 0.f;
        #pragma unroll
        for (int j = 0; j < 4; ++j) { float d = y[j] - mu2; q2 += d * d; }
        q2 = wave_sum(q2); float rs2 = rsqrtf(q2 * (1.f / 128.f) + 1e-5f);
        #pragma unroll
        for (int j = 0; j < 4; ++j) { int c = lane + 32 * j; tile[c][wl] = (y[j] - mu2) * rs2 * gF[c] + bF[c]; }
    }
    __syncthreads();
    #pragma unroll
    for (int it = 0; it < 16; ++it) {
        int cc = it * 8 + wv;
        out[(((size_t)b * 128 + cc) * 256 + h) * 256 + wch * 32 + lane] = tile[cc][lane];
    }
}

// ============================ launch ============================
extern "C" void kernel_launch(void* const* d_in, const int* in_sizes, int n_in,
                              void* d_out, int out_size, void* d_ws, size_t ws_size,
                              hipStream_t stream) {
    (void)in_sizes; (void)n_in; (void)out_size; (void)ws_size;
    // inputs: 0 rgb_feat, 1 hsi_feat, then params tree-flattened (dict keys sorted)
    const float* rgb = (const float*)d_in[0];
    const float* hsi = (const float*)d_in[1];
    const float* alpha   = (const float*)d_in[2];
    const float* fuse_g  = (const float*)d_in[3];
    const float* fuse_b  = (const float*)d_in[4];
    const float* h2rk_w  = (const float*)d_in[5];  const float* h2rk_b = (const float*)d_in[6];
    const float* h2rp_w  = (const float*)d_in[7];  const float* h2rp_b = (const float*)d_in[8];
    const float* h2rq_w  = (const float*)d_in[9];  const float* h2rq_b = (const float*)d_in[10];
    const float* h2rv_w  = (const float*)d_in[11]; const float* h2rv_b = (const float*)d_in[12];
    const float* hffn_w1 = (const float*)d_in[13]; const float* hffn_b1 = (const float*)d_in[14];
    const float* hffn_w2 = (const float*)d_in[15]; const float* hffn_b2 = (const float*)d_in[16];
    const float* lnh1_g = (const float*)d_in[17]; const float* lnh1_b = (const float*)d_in[18];
    const float* lnh2_g = (const float*)d_in[19]; const float* lnh2_b = (const float*)d_in[20];
    const float* lnhf_g = (const float*)d_in[21]; const float* lnhf_b = (const float*)d_in[22];
    const float* lnr1_g = (const float*)d_in[23]; const float* lnr1_b = (const float*)d_in[24];
    const float* lnr2_g = (const float*)d_in[25]; const float* lnr2_b = (const float*)d_in[26];
    const float* lnrf_g = (const float*)d_in[27]; const float* lnrf_b = (const float*)d_in[28];
    const float* r2hk_w = (const float*)d_in[29]; const float* r2hk_b = (const float*)d_in[30];
    const float* r2hp_w = (const float*)d_in[31]; const float* r2hp_b = (const float*)d_in[32];
    const float* r2hq_w = (const float*)d_in[33]; const float* r2hq_b = (const float*)d_in[34];
    const float* r2hv_w = (const float*)d_in[35]; const float* r2hv_b = (const float*)d_in[36];
    const float* rffn_w1 = (const float*)d_in[37]; const float* rffn_b1 = (const float*)d_in[38];
    const float* rffn_w2 = (const float*)d_in[39]; const float* rffn_b2 = (const float*)d_in[40];
    const float* se_w1 = (const float*)d_in[41]; const float* se_b1 = (const float*)d_in[42];
    const float* se_w2 = (const float*)d_in[43]; const float* se_b2 = (const float*)d_in[44];
    float* out = (float*)d_out;

    // workspace layout (bytes, 256-aligned); qkv region later aliased by f32 residual streams
    char* base = (char*)d_ws;
    bf16_t* WT = (bf16_t*)base;                               // 786432 B of bf16 weights
    float* partials = (float*)(base + 786432);                // 496*128*4
    float* wbuf     = (float*)(base + 1040384);               // 2*128*4
    float* FUS      = (float*)(base + 1041408);               // 8192*128*4
    bf16_t* lnr     = (bf16_t*)(base + 5235712);              // Nr*128 bf16 (reused as attnR)
    bf16_t* attnR   = lnr;
    bf16_t* lnh     = (bf16_t*)(base + 38790144);             // Nh*128 bf16 (reused as attnH)
    bf16_t* attnH   = lnh;
    char* E = base + 103801856;
    bf16_t* qc = (bf16_t*)(E);
    bf16_t* kc = (bf16_t*)(E + 33554432);
    bf16_t* vc = (bf16_t*)(E + 67108864);
    bf16_t* kh = (bf16_t*)(E + 100663296);
    bf16_t* vh = (bf16_t*)(E + 165675008);
    bf16_t* qh = (bf16_t*)(E + 230686720);
    float* zR = (float*)(E);                                  // aliases qc+kc after attention
    float* zH = (float*)(E + 100663296);                      // aliases kh+vh after attention

    bf16_t* t_r2hq = WT + 0 * 16384; bf16_t* t_r2hk = WT + 1 * 16384;
    bf16_t* t_r2hv = WT + 2 * 16384; bf16_t* t_r2hp = WT + 3 * 16384;
    bf16_t* t_h2rq = WT + 4 * 16384; bf16_t* t_h2rk = WT + 5 * 16384;
    bf16_t* t_h2rv = WT + 6 * 16384; bf16_t* t_h2rp = WT + 7 * 16384;
    bf16_t* t_rw1 = WT + 8 * 16384;  bf16_t* t_rw2 = t_rw1 + 65536;
    bf16_t* t_hw1 = t_rw2 + 65536;   bf16_t* t_hw2 = t_hw1 + 65536;

    const int NR_TILES = 131072 / 16;   // 8192
    const int NH_TILES = 253952 / 16;   // 15872

    // 1) weights -> bf16 transposed [N][K]
    k_transpose_bf16<<<64, 256, 0, stream>>>(r2hq_w, t_r2hq, 128, 128);
    k_transpose_bf16<<<64, 256, 0, stream>>>(r2hk_w, t_r2hk, 128, 128);
    k_transpose_bf16<<<64, 256, 0, stream>>>(r2hv_w, t_r2hv, 128, 128);
    k_transpose_bf16<<<64, 256, 0, stream>>>(r2hp_w, t_r2hp, 128, 128);
    k_transpose_bf16<<<64, 256, 0, stream>>>(h2rq_w, t_h2rq, 128, 128);
    k_transpose_bf16<<<64, 256, 0, stream>>>(h2rk_w, t_h2rk, 128, 128);
    k_transpose_bf16<<<64, 256, 0, stream>>>(h2rv_w, t_h2rv, 128, 128);
    k_transpose_bf16<<<64, 256, 0, stream>>>(h2rp_w, t_h2rp, 128, 128);
    k_transpose_bf16<<<256, 256, 0, stream>>>(rffn_w1, t_rw1, 128, 512);
    k_transpose_bf16<<<256, 256, 0, stream>>>(rffn_w2, t_rw2, 512, 128);
    k_transpose_bf16<<<256, 256, 0, stream>>>(hffn_w1, t_hw1, 128, 512);
    k_transpose_bf16<<<256, 256, 0, stream>>>(hffn_w2, t_hw2, 512, 128);

    // 2) LayerNorms + layout transforms
    k_ln_rgb1<<<4096, 256, 0, stream>>>(rgb, lnr1_g, lnr1_b, lnr);
    k_ln_hsi1<<<7936, 256, 0, stream>>>(hsi, lnh1_g, lnh1_b, lnh);

    // 3) q/k/v projections (WMMA)
    k_gemm_bias<<<NR_TILES, 256, 0, stream>>>(lnr, t_r2hq, r2hq_b, qc);
    k_gemm_bias<<<NR_TILES, 256, 0, stream>>>(lnr, t_h2rk, h2rk_b, kc);
    k_gemm_bias<<<NR_TILES, 256, 0, stream>>>(lnr, t_h2rv, h2rv_b, vc);
    k_gemm_bias<<<NH_TILES, 256, 0, stream>>>(lnh, t_r2hk, r2hk_b, kh);
    k_gemm_bias<<<NH_TILES, 256, 0, stream>>>(lnh, t_r2hv, r2hv_b, vh);
    k_gemm_bias<<<NH_TILES, 256, 0, stream>>>(lnh, t_h2rq, h2rq_b, qh);

    // 4) cross-attention (overwrites ln buffers)
    k_attn_r2h<<<8192, 256, 0, stream>>>(qc, kh, vh, attnR);
    k_attn_h2r<<<8192, 256, 0, stream>>>(qh, kc, vc, attnH);

    // 5) output projections + residuals (overwrite qkv region with f32 streams)
    k_proj_res_rgb<<<NR_TILES, 256, 0, stream>>>(attnR, t_r2hp, r2hp_b, rgb, zR);
    k_proj_res_hsi<<<NH_TILES, 256, 0, stream>>>(attnH, t_h2rp, h2rp_b, hsi, zH);

    // 6) FFNs (LN + GEMM + GELU + GEMM + residual), in place
    k_ffn_fused<<<NR_TILES, 128, 0, stream>>>(zR, lnr2_g, lnr2_b, t_rw1, rffn_b1, t_rw2, rffn_b2);
    k_ffn_fused<<<NH_TILES, 128, 0, stream>>>(zH, lnh2_g, lnh2_b, t_hw1, hffn_b1, t_hw2, hffn_b2);

    // 7) SpectralSE
    k_se_pool<<<496, 128, 0, stream>>>(zH, partials);
    k_se_mlp<<<1, 128, 0, stream>>>(partials, se_w1, se_b1, se_w2, se_b2, wbuf);
    k_collapse<<<1024, 256, 0, stream>>>(zH, wbuf, lnhf_g, lnhf_b, FUS);

    // 8) final fuse + transpose to (B,C,H,W)
    k_final<<<4096, 256, 0, stream>>>(zR, FUS, alpha, lnrf_g, lnrf_b, fuse_g, fuse_b, out);
}
